// TranslationInvariantMP_52785148067991
// MI455X (gfx1250) — matrix operand order repaired and checked by reading, not compiled, not run on hardware
//
#include <hip/hip_runtime.h>

// TranslationInvariantMP for MI455X (gfx1250, wave32).
// V=100000 vertices, K=32 neighbours, F=64 features, two dense 64x64 layers.
// Gather phase: per-lane full-row f32 accumulation (b128 loads).
// Dense phase: v_wmma_f32_16x16x32_f16, f32 accumulate, fragments built per
// the CDNA5 ISA 16-bit A/B layouts and 32-bit C/D layout.

#define V_TOT 100000
#define KNN   32
#define FDIM  64
#define OUTW  128   // concat(h0, h1)

typedef __attribute__((ext_vector_type(16))) _Float16 v16h;
typedef __attribute__((ext_vector_type(8)))  float    v8f;

template<bool FIRST>
__global__ __launch_bounds__(128)
void tmp_mp_kernel(const float* __restrict__ feat,   // layer input rows
                   long featStride,                  // 64 for x, 128 for h0-in-out
                   const int*   __restrict__ nbr,    // [V,K]
                   const float* __restrict__ dsq,    // [V,K]
                   const float* __restrict__ W,      // [64,64] row-major (K x N)
                   const float* __restrict__ bias,   // [64] or unused
                   float* __restrict__ out,          // [V,128]
                   int colOff)                       // 0 for h0, 64 for h1
{
    // per-wave f16 staging tile: 32 vertices x 64 features (+pad vs bank conflicts)
    __shared__ _Float16 tileA[4][32][FDIM + 8];

    const int wave = threadIdx.x >> 5;
    const int lane = threadIdx.x & 31;
    const long vbase = ((long)blockIdx.x * 4 + wave) * 32;
    const bool active = vbase < V_TOT;   // wave-uniform: V % 32 == 0

    if (active) {
        const long v = vbase + lane;     // lane owns one vertex
        float acc[FDIM];
        #pragma unroll
        for (int f = 0; f < FDIM; ++f) acc[f] = 0.0f;

        float wsum = 0.0f;
        const int*   nrow = nbr + v * KNN;
        const float* drow = dsq + v * KNN;

        for (int k = 0; k < KNN; ++k) {
            const float w = __expf(-10.0f * drow[k]);
            wsum += w;
            const long j = (long)nrow[k];
            const float4* xr = (const float4*)(feat + j * featStride);
            #pragma unroll
            for (int q = 0; q < FDIM / 4; ++q) {
                const float4 t = xr[q];                 // 256B contiguous per lane
                acc[4*q+0] += w * t.x;
                acc[4*q+1] += w * t.y;
                acc[4*q+2] += w * t.z;
                acc[4*q+3] += w * t.w;
            }
        }

        if (FIRST) {
            // translation invariance: acc -= x_i * sum_k w
            const float4* xv = (const float4*)(feat + v * featStride);
            #pragma unroll
            for (int q = 0; q < FDIM / 4; ++q) {
                const float4 t = xv[q];
                acc[4*q+0] -= wsum * t.x;
                acc[4*q+1] -= wsum * t.y;
                acc[4*q+2] -= wsum * t.z;
                acc[4*q+3] -= wsum * t.w;
            }
        }

        const float inv = 1.0f / (float)KNN;
        #pragma unroll
        for (int f = 0; f < FDIM; ++f)
            tileA[wave][lane][f] = (_Float16)(acc[f] * inv);
    }

    __syncthreads();
    if (!active) return;

    // ---- dense layer via WMMA: [32 x 64] @ [64 x 64] ----
    const int m  = lane & 15;   // row within M-tile (A) / column N (B, C/D)
    const int hs = lane >> 4;   // lane half selector

    // A fragments: two 16x32 tiles per K-half.
    // ISA 16-bit A 16x32 layout: VGPR p holds K = (p<4?0:16) + hs*8 + (p&3)*2, +1
    v16h a[2][2];
    #pragma unroll
    for (int mt = 0; mt < 2; ++mt)
        #pragma unroll
        for (int kt = 0; kt < 2; ++kt)
            #pragma unroll
            for (int p = 0; p < 8; ++p) {
                const int k0 = kt * 32 + ((p < 4) ? 0 : 16) + hs * 8 + (p & 3) * 2;
                a[mt][kt][2*p]   = tileA[wave][mt*16 + m][k0];
                a[mt][kt][2*p+1] = tileA[wave][mt*16 + m][k0 + 1];
            }

    for (int nt = 0; nt < 4; ++nt) {
        // B fragments: 32x16 tile; lane holds column n, VGPR p holds
        // K = (p<4?0:16) + hs*8 + (p&3)*2, +1 (row-major W, stride 64)
        const int n = nt * 16 + m;
        v16h b[2];
        #pragma unroll
        for (int kt = 0; kt < 2; ++kt)
            #pragma unroll
            for (int p = 0; p < 8; ++p) {
                const int k0 = kt * 32 + ((p < 4) ? 0 : 16) + hs * 8 + (p & 3) * 2;
                b[kt][2*p]   = (_Float16)W[(long)k0 * FDIM + n];
                b[kt][2*p+1] = (_Float16)W[(long)(k0 + 1) * FDIM + n];
            }

        const float bb = FIRST ? 0.0f : bias[n];

        #pragma unroll
        for (int mt = 0; mt < 2; ++mt) {
            v8f c = {};
            c = __builtin_amdgcn_wmma_f32_16x16x32_f16(
                    false, a[mt][0], false, b[0], (short)0, c, false, false);
            c = __builtin_amdgcn_wmma_f32_16x16x32_f16(
                    false, a[mt][1], false, b[1], (short)0, c, false, false);

            // C/D layout: VGPR r -> M = hs*8 + r, N = m
            #pragma unroll
            for (int r = 0; r < 8; ++r) {
                const float z = c[r] + bb;
                const float e = (z > 0.0f) ? z : (__expf(z) - 1.0f);  // ELU
                const long row = vbase + mt * 16 + hs * 8 + r;
                out[row * OUTW + colOff + n] = e;
            }
        }
    }
}

extern "C" void kernel_launch(void* const* d_in, const int* in_sizes, int n_in,
                              void* d_out, int out_size, void* d_ws, size_t ws_size,
                              hipStream_t stream) {
    (void)in_sizes; (void)n_in; (void)d_ws; (void)ws_size; (void)out_size;

    const float* x   = (const float*)d_in[0];
    const int*   nbr = (const int*)  d_in[1];
    const float* dsq = (const float*)d_in[2];
    const float* W0  = (const float*)d_in[3];
    const float* W1  = (const float*)d_in[4];
    const float* b1  = (const float*)d_in[5];
    float* out = (float*)d_out;

    const int numWaves = V_TOT / 32;                // 3125
    const dim3 grid((numWaves + 3) / 4);            // 4 waves / block
    const dim3 block(128);

    // Layer 0: aggregate x (translation-invariant) -> @W0 -> ELU -> out[:, 0:64]
    tmp_mp_kernel<true><<<grid, block, 0, stream>>>(
        x, (long)FDIM, nbr, dsq, W0, (const float*)nullptr, out, 0);

    // Layer 1: aggregate h0 (read from out cols 0:64) -> @W1+b1 -> ELU -> out[:, 64:128]
    tmp_mp_kernel<false><<<grid, block, 0, stream>>>(
        out, (long)OUTW, nbr, dsq, W1, b1, out, FDIM);
}
